// UnfusedGemma4TextExperts_43215960933136
// MI455X (gfx1250) — matrix operand order repaired and checked by reading, not compiled, not run on hardware
//
#include <hip/hip_runtime.h>
#include <hip/hip_bf16.h>

// MoE GeGLU FFN for MI455X (gfx1250), bf16 WMMA path, v2.1.
// Pre-converted bf16 weights, BM=256 x BN=128 tiles with 512-thread blocks,
// LDS ping-pong double buffering fed by gfx1250 async global->LDS DMA
// (ASYNCcnt). Workspace requirement: ~571 MB.

#define NE 8
#define NT 8192
#define NK 2
#define NH 2048
#define NI 4096
#define NTK (NT * NK)

#define BM 256        // token rows per block tile
#define BN 128        // output cols per block tile
#define BK 32         // k-slice
#define LDSS 40       // padded LDS row stride (elements); 80B rows keep 16B align

typedef __bf16 bf16;
typedef __attribute__((ext_vector_type(16))) __bf16 v16bf;
typedef __attribute__((ext_vector_type(8)))  __bf16 v8bf;
typedef __attribute__((ext_vector_type(4)))  __bf16 v4bf;
typedef __attribute__((ext_vector_type(8)))  float   v8f;
typedef __attribute__((ext_vector_type(4)))  float   v4f;
typedef int v4i __attribute__((vector_size(16)));

#if __has_builtin(__builtin_amdgcn_global_load_async_to_lds_b128) && \
    __has_builtin(__builtin_amdgcn_s_wait_asynccnt)
#define USE_ASYNC 1
#endif

// 16-byte global->LDS copy: async DMA on gfx1250, register-staged fallback.
__device__ __forceinline__ void cp16(const bf16* g, bf16* l) {
#ifdef USE_ASYNC
  __builtin_amdgcn_global_load_async_to_lds_b128(
      (__attribute__((address_space(1))) v4i*)g,
      (__attribute__((address_space(3))) v4i*)l, 0, 0);
#else
  *(v8bf*)l = *(const v8bf*)g;
#endif
}

#ifdef USE_ASYNC
#define WAIT_ASYNC(n) __builtin_amdgcn_s_wait_asynccnt(n)
#else
#define WAIT_ASYNC(n)
#endif

__device__ __forceinline__ v16bf cat8(v8bf lo, v8bf hi) {
  return __builtin_shufflevector(lo, hi, 0, 1, 2, 3, 4, 5, 6, 7,
                                         8, 9, 10, 11, 12, 13, 14, 15);
}

__device__ __forceinline__ float gelu_tanh(float x) {
  float x3 = x * x * x;
  return 0.5f * x * (1.0f + tanhf(0.7978845608028654f * (x + 0.044715f * x3)));
}

// ---------------- routing / setup kernels ----------------

__global__ void k_init_counts(int* counts_cursors) {
  int i = threadIdx.x;
  if (i < 2 * NE) counts_cursors[i] = 0;
}

__global__ void k_zero_f32(float* p, long n) {
  long i = (long)blockIdx.x * blockDim.x + threadIdx.x;
  long stride = (long)gridDim.x * blockDim.x;
  for (; i < n; i += stride) p[i] = 0.0f;
}

__global__ void k_count(const int* __restrict__ idx, int* __restrict__ counts) {
  int i = blockIdx.x * blockDim.x + threadIdx.x;
  if (i < NTK) atomicAdd(&counts[idx[i]], 1);
}

__global__ void k_scan(const int* __restrict__ counts, int* __restrict__ offsets) {
  if (threadIdx.x == 0) {
    int acc = 0;
    for (int e = 0; e < NE; ++e) { offsets[e] = acc; acc += counts[e]; }
  }
}

__global__ void k_assign(const int* __restrict__ idx, const float* __restrict__ w,
                         const int* __restrict__ offsets, int* __restrict__ cursors,
                         int* __restrict__ token_ids, float* __restrict__ combine) {
  int i = blockIdx.x * blockDim.x + threadIdx.x;
  if (i < NTK) {
    int e = idx[i];
    int p = atomicAdd(&cursors[e], 1);
    int s = offsets[e] + p;
    token_ids[s] = i / NK;
    combine[s] = w[i];
  }
}

// Vectorized fp32 -> bf16 conversion (n must be a multiple of 4).
__global__ void k_cvt4_bf16(const float* __restrict__ x, bf16* __restrict__ y, long n4) {
  long i = (long)blockIdx.x * blockDim.x + threadIdx.x;
  long stride = (long)gridDim.x * blockDim.x;
  for (; i < n4; i += stride) {
    v4f v = ((const v4f*)x)[i];
    v4bf o;
#pragma unroll
    for (int q = 0; q < 4; ++q) o[q] = (bf16)v[q];
    ((v4bf*)y)[i] = o;
  }
}

// ---------------- GEMM1: act = gelu(X @ Gw^T) * (X @ Uw^T) ----------------

__global__ __launch_bounds__(512)
void k_gemm1(const bf16* __restrict__ X, const bf16* __restrict__ gw,
             const bf16* __restrict__ uw, const int* __restrict__ token_ids,
             const int* __restrict__ counts, const int* __restrict__ offsets,
             bf16* __restrict__ act) {
  __shared__ bf16 sA[2][BM * LDSS];
  __shared__ bf16 sG[2][BN * LDSS];
  __shared__ bf16 sU[2][BN * LDSS];
  __shared__ int sTok[BM];

  const int e = blockIdx.z;
  const int cnt = counts[e];
  const int off = offsets[e];
  const int i0 = blockIdx.x * BN;
  const int tid = (int)threadIdx.x;
  const int lane = tid & 31;
  const int wave = tid >> 5;      // 16 waves
  const int wm = wave & 3;        // 4 wave rows of 64
  const int wn = wave >> 2;       // 4 wave cols of 32
  const int lhalf = lane >> 4;
  const int lmod = lane & 15;
  const int mtiles = (cnt + BM - 1) / BM;
  const int KT = NH / BK;  // 64

  const bf16* gbase = gw + (size_t)e * NI * NH;
  const bf16* ubase = uw + (size_t)e * NI * NH;

  // Per-thread B-tile copy descriptor: 128 rows x 4 chunks -> 1 chunk/thread.
  const int br = tid >> 2, bc = tid & 3;
  const bf16* gSrc = gbase + (size_t)(i0 + br) * NH + bc * 8;
  const bf16* uSrc = ubase + (size_t)(i0 + br) * NH + bc * 8;
  const int bDst = br * LDSS + bc * 8;

  for (int mt = blockIdx.y; mt < mtiles; mt += gridDim.y) {
    const int mBase = mt * BM;
    if (tid < BM) {
      int s = mBase + tid;
      // Clamp (not mask): padding rows load token 0's data; results are
      // discarded in the guarded epilogue. Keeps async issue-count uniform.
      sTok[tid] = (s < cnt) ? token_ids[off + s] : 0;
    }
    __syncthreads();

    // Per-thread A-tile copy descriptors: 256 rows x 4 chunks -> 2/thread.
    const bf16* aSrc[2];
    int aDst[2];
#pragma unroll
    for (int j = 0; j < 2; ++j) {
      int cid = tid + j * 512;
      int r = cid >> 2, c = cid & 3;
      aSrc[j] = X + (size_t)sTok[r] * NH + c * 8;
      aDst[j] = r * LDSS + c * 8;
    }

    v8f accG[4][2];
    v8f accU[4][2];
#pragma unroll
    for (int f = 0; f < 4; ++f)
#pragma unroll
      for (int g = 0; g < 2; ++g) { accG[f][g] = (v8f)0.0f; accU[f][g] = (v8f)0.0f; }

    // Prologue: DMA tile 0 into buffer 0. (4 async ops per wave per step.)
#pragma unroll
    for (int j = 0; j < 2; ++j) cp16(aSrc[j], &sA[0][aDst[j]]);
    cp16(gSrc, &sG[0][bDst]);
    cp16(uSrc, &sU[0][bDst]);

    for (int kt = 0; kt < KT; ++kt) {
      const int cur = kt & 1, nxt = cur ^ 1;
      if (kt + 1 < KT) {
        const int k1 = (kt + 1) * BK;
#pragma unroll
        for (int j = 0; j < 2; ++j) cp16(aSrc[j] + k1, &sA[nxt][aDst[j]]);
        cp16(gSrc + k1, &sG[nxt][bDst]);
        cp16(uSrc + k1, &sU[nxt][bDst]);
        WAIT_ASYNC(4);   // tile kt complete; kt+1 still in flight
      } else {
        WAIT_ASYNC(0);
      }
      __syncthreads();   // all waves' tile-kt DMA complete

      v16bf af[4];
#pragma unroll
      for (int f = 0; f < 4; ++f) {
        const bf16* ap = &sA[cur][(wm * 64 + f * 16 + lmod) * LDSS + 16 * lhalf];
        af[f] = cat8(*(const v8bf*)ap, *(const v8bf*)(ap + 8));
      }
#pragma unroll
      for (int g = 0; g < 2; ++g) {
        const bf16* gp = &sG[cur][(wn * 32 + g * 16 + lmod) * LDSS + 16 * lhalf];
        v16bf bg = cat8(*(const v8bf*)gp, *(const v8bf*)(gp + 8));
        const bf16* up = &sU[cur][(wn * 32 + g * 16 + lmod) * LDSS + 16 * lhalf];
        v16bf bu = cat8(*(const v8bf*)up, *(const v8bf*)(up + 8));
#pragma unroll
        for (int f = 0; f < 4; ++f) {
          accG[f][g] = __builtin_amdgcn_wmma_f32_16x16x32_bf16(
              false, af[f], false, bg, (short)0, accG[f][g], false, false);
          accU[f][g] = __builtin_amdgcn_wmma_f32_16x16x32_bf16(
              false, af[f], false, bu, (short)0, accU[f][g], false, false);
        }
      }
      __syncthreads();   // all waves done reading buf[cur]
    }

    // Epilogue: h = gelu(g) * u -> bf16 act
#pragma unroll
    for (int f = 0; f < 4; ++f)
#pragma unroll
      for (int g = 0; g < 2; ++g)
#pragma unroll
        for (int r = 0; r < 8; ++r) {
          int s = mBase + wm * 64 + f * 16 + r + 8 * lhalf;
          if (s < cnt) {
            int col = i0 + wn * 32 + g * 16 + lmod;
            float h = gelu_tanh(accG[f][g][r]) * accU[f][g][r];
            act[(size_t)(off + s) * NI + col] = (bf16)h;
          }
        }
    __syncthreads();
  }
}

// ---------------- GEMM2: out += combine * (act @ Dw^T) ----------------

__global__ __launch_bounds__(512)
void k_gemm2(const bf16* __restrict__ act, const bf16* __restrict__ dw,
             const int* __restrict__ token_ids, const float* __restrict__ combine,
             const int* __restrict__ counts, const int* __restrict__ offsets,
             float* __restrict__ out) {
  __shared__ bf16 sA[2][BM * LDSS];
  __shared__ bf16 sB[2][BN * LDSS];

  const int e = blockIdx.z;
  const int cnt = counts[e];
  const int off = offsets[e];
  const int h0 = blockIdx.x * BN;   // over NH (16 tiles)
  const int tid = (int)threadIdx.x;
  const int lane = tid & 31;
  const int wave = tid >> 5;
  const int wm = wave & 3;
  const int wn = wave >> 2;
  const int lhalf = lane >> 4;
  const int lmod = lane & 15;
  const int mtiles = (cnt + BM - 1) / BM;
  const int KT = NI / BK;  // 128

  const bf16* dbase = dw + (size_t)e * NH * NI;
  const int br = tid >> 2, bc = tid & 3;
  const bf16* bSrc = dbase + (size_t)(h0 + br) * NI + bc * 8;
  const int bDst = br * LDSS + bc * 8;

  for (int mt = blockIdx.y; mt < mtiles; mt += gridDim.y) {
    const int mBase = mt * BM;

    const bf16* aSrc[2];
    int aDst[2];
#pragma unroll
    for (int j = 0; j < 2; ++j) {
      int cid = tid + j * 512;
      int r = cid >> 2, c = cid & 3;
      long row = (long)off + mBase + r;
      if (row > (long)NTK - 1) row = (long)NTK - 1;  // clamp, discard later
      aSrc[j] = act + (size_t)row * NI + c * 8;
      aDst[j] = r * LDSS + c * 8;
    }

    v8f acc[4][2];
#pragma unroll
    for (int f = 0; f < 4; ++f)
#pragma unroll
      for (int g = 0; g < 2; ++g) acc[f][g] = (v8f)0.0f;

    // Prologue: DMA tile 0. (3 async ops per wave per step.)
#pragma unroll
    for (int j = 0; j < 2; ++j) cp16(aSrc[j], &sA[0][aDst[j]]);
    cp16(bSrc, &sB[0][bDst]);

    for (int kt = 0; kt < KT; ++kt) {
      const int cur = kt & 1, nxt = cur ^ 1;
      if (kt + 1 < KT) {
        const int k1 = (kt + 1) * BK;
#pragma unroll
        for (int j = 0; j < 2; ++j) cp16(aSrc[j] + k1, &sA[nxt][aDst[j]]);
        cp16(bSrc + k1, &sB[nxt][bDst]);
        WAIT_ASYNC(3);
      } else {
        WAIT_ASYNC(0);
      }
      __syncthreads();

      v16bf af[4];
#pragma unroll
      for (int f = 0; f < 4; ++f) {
        const bf16* ap = &sA[cur][(wm * 64 + f * 16 + lmod) * LDSS + 16 * lhalf];
        af[f] = cat8(*(const v8bf*)ap, *(const v8bf*)(ap + 8));
      }
#pragma unroll
      for (int g = 0; g < 2; ++g) {
        const bf16* bp = &sB[cur][(wn * 32 + g * 16 + lmod) * LDSS + 16 * lhalf];
        v16bf b = cat8(*(const v8bf*)bp, *(const v8bf*)(bp + 8));
#pragma unroll
        for (int f = 0; f < 4; ++f) {
          acc[f][g] = __builtin_amdgcn_wmma_f32_16x16x32_bf16(
              false, af[f], false, b, (short)0, acc[f][g], false, false);
        }
      }
      __syncthreads();
    }

    // Epilogue: scale by combine weight and scatter-add into out
#pragma unroll
    for (int f = 0; f < 4; ++f)
#pragma unroll
      for (int g = 0; g < 2; ++g)
#pragma unroll
        for (int r = 0; r < 8; ++r) {
          int s = mBase + wm * 64 + f * 16 + r + 8 * lhalf;
          if (s < cnt) {
            int srow = off + s;
            int tok = token_ids[srow];
            float w = combine[srow];
            int col = h0 + wn * 32 + g * 16 + lmod;
            atomicAdd(&out[(size_t)tok * NH + col], w * acc[f][g][r]);
          }
        }
    __syncthreads();
  }
}

// ---------------- host launcher ----------------

extern "C" void kernel_launch(void* const* d_in, const int* in_sizes, int n_in,
                              void* d_out, int out_size, void* d_ws, size_t ws_size,
                              hipStream_t stream) {
  (void)in_sizes; (void)n_in; (void)out_size; (void)ws_size;
  const float* hs  = (const float*)d_in[0];
  const float* tkw = (const float*)d_in[1];
  const float* gw  = (const float*)d_in[2];
  const float* uw  = (const float*)d_in[3];
  const float* dw  = (const float*)d_in[4];
  const int*   idx = (const int*)d_in[5];
  float* out = (float*)d_out;

  // Workspace carve-up (~571 MB total)
  char* ws = (char*)d_ws;
  int* counts  = (int*)ws;        // [NE]
  int* cursors = counts + NE;     // [NE]
  int* offsets = cursors + NE;    // [NE]
  size_t o = 256;
  int*   token_ids = (int*)(ws + o);   o += (size_t)NTK * 4;
  float* combine   = (float*)(ws + o); o += (size_t)NTK * 4;
  bf16*  Xb  = (bf16*)(ws + o);        o += (size_t)NT * NH * 2;
  bf16*  actb = (bf16*)(ws + o);       o += (size_t)NTK * NI * 2;
  bf16*  gwb = (bf16*)(ws + o);        o += (size_t)NE * NI * NH * 2;
  bf16*  uwb = (bf16*)(ws + o);        o += (size_t)NE * NI * NH * 2;
  bf16*  dwb = (bf16*)(ws + o);        o += (size_t)NE * NH * NI * 2;

  k_init_counts<<<1, 64, 0, stream>>>(counts);
  k_zero_f32<<<4096, 256, 0, stream>>>(out, (long)NT * NH);
  k_count<<<NTK / 256, 256, 0, stream>>>(idx, counts);
  k_scan<<<1, 32, 0, stream>>>(counts, offsets);
  k_assign<<<NTK / 256, 256, 0, stream>>>(idx, tkw, offsets, cursors, token_ids, combine);

  k_cvt4_bf16<<<4096, 256, 0, stream>>>(hs, Xb, (long)NT * NH / 4);
  k_cvt4_bf16<<<8192, 256, 0, stream>>>(gw, gwb, (long)NE * NI * NH / 4);
  k_cvt4_bf16<<<8192, 256, 0, stream>>>(uw, uwb, (long)NE * NI * NH / 4);
  k_cvt4_bf16<<<8192, 256, 0, stream>>>(dw, dwb, (long)NE * NH * NI / 4);

  k_gemm1<<<dim3(NI / BN, 8, NE), 512, 0, stream>>>(Xb, gwb, uwb, token_ids, counts, offsets, actb);
  k_gemm2<<<dim3(NH / BN, 8, NE), 512, 0, stream>>>(actb, dwb, token_ids, combine, counts, offsets, out);
}
